// VRWKV_SpatialMix_V6_85066122264778
// MI455X (gfx1250) — matrix operand order repaired and checked by reading, not compiled, not run on hardware
//
#include <hip/hip_runtime.h>
#include <hip/hip_bf16.h>

// ---------------------------------------------------------------------------
// VRWKV SpatialMix V6 for gfx1250 (MI455X)
// bf16 WMMA GEMM pipeline (LDS-staged B via async-to-LDS, double buffered)
// + software-pipelined wkv6 scan + fused epilogues (hw tanh).
// ---------------------------------------------------------------------------

typedef __attribute__((ext_vector_type(16))) __bf16 v16bf;
typedef __attribute__((ext_vector_type(8)))  __bf16 v8bf;
typedef __attribute__((ext_vector_type(8)))  float  v8f;

#define Bn   4
#define Tn   1024
#define Cd   768
#define NHn  24
#define HSn  32
#define Mrows (Bn * Tn)          // 4096

// Epilogue selectors
#define EPI_F32       0
#define EPI_TANH_BF16 1
#define EPI_MIX_BF16  2
#define EPI_SILU_F32  3
#define EPI_DECAY_F32 4

// CDNA5 has a hardware V_TANH_F32 transcendental; use it if clang exposes it.
__device__ __forceinline__ float fast_tanh(float x) {
#if __has_builtin(__builtin_amdgcn_tanhf)
    return __builtin_amdgcn_tanhf(x);
#else
    float e = __expf(2.0f * x);          // v_exp_f32 based, branch-free
    return 1.0f - 2.0f / (e + 1.0f);
#endif
}

// ---------------------------------------------------------------------------
// Weight transpose + fp32 -> bf16 convert.  src[R][Cc] f32  ->  dst[Cpad][R] bf16
// ---------------------------------------------------------------------------
__global__ __launch_bounds__(256)
void transpose_bf16_kernel(const float* __restrict__ src, __bf16* __restrict__ dst,
                           int R, int Cc, int Cpad) {
    int idx = blockIdx.x * 256 + threadIdx.x;
    if (idx >= R * Cpad) return;
    int r = idx % R;
    int c = idx / R;
    float v = (c < Cc) ? src[(size_t)r * Cc + c] : 0.0f;
    dst[(size_t)c * R + r] = (__bf16)v;
}

// ---------------------------------------------------------------------------
// q_shift + time_maa_x mixing.
// ---------------------------------------------------------------------------
__global__ __launch_bounds__(256)
void prep_kernel(const float* __restrict__ x, const float* __restrict__ tmx,
                 float* __restrict__ xx, __bf16* __restrict__ xmix) {
    int idx = blockIdx.x * 256 + threadIdx.x;
    if (idx >= Mrows * Cd) return;
    int c = idx % Cd;
    int t = (idx / Cd) % Tn;
    int b = idx / (Cd * Tn);
    int h = t >> 5, w = t & 31;

    int q = c / (Cd / 4);
    int hh = h, ww = w; bool ok;
    if      (q == 0) { ww = w - 1; ok = (w > 0);  }
    else if (q == 1) { ww = w + 1; ok = (w < 31); }
    else if (q == 2) { hh = h - 1; ok = (h > 0);  }
    else             { hh = h + 1; ok = (h < 31); }

    float sv = 0.0f;
    if (ok) sv = x[((size_t)b * Tn + hh * 32 + ww) * Cd + c];
    float xv = x[idx];
    float d  = sv - xv;
    xx[idx]  = d;
    xmix[idx] = (__bf16)(xv + d * tmx[c]);
}

// ---------------------------------------------------------------------------
// WMMA GEMM:  C[M][N] = A[M][K](bf16,row-major,lda) * Bt[N][K](bf16) +epilogue
// K % 32 == 0, N % 64 == 0, M % 256 == 0.
//
// Block = 256 thr = 8 waves.  Block tile: 256 (M) x 64 (N).
// Wave tile: 32 (M) x 64 (N) = 2 A-frags x 4 B-tiles = 8 wmmas / K-step.
// B tile staged in LDS via global_load_async_to_lds_b128 (double buffered,
// s_wait_asynccnt + barrier).  A frags register double-buffered.  All four
// B fragments are pulled from LDS into regs before the 8-wmma burst.
// ---------------------------------------------------------------------------
template <int EPI>
__global__ __launch_bounds__(256)
void gemm_wmma_kernel(const __bf16* __restrict__ A, int lda,
                      const __bf16* __restrict__ Bt, int K,
                      void* __restrict__ Cout, int ldc,
                      const float* __restrict__ e_x,
                      const float* __restrict__ e_xx,
                      const float* __restrict__ e_vec) {
    __shared__ __align__(16) __bf16 ldsB[2][64 * 32];   // 2 x 4KB

    const int tid  = threadIdx.x;
    const int lane = tid & 31;
    const int wave = tid >> 5;
    const int m    = lane & 15;
    const int kh   = lane >> 4;
    const int row0 = blockIdx.y * 256 + wave * 32;
    const int n0   = blockIdx.x * 64;

    // cooperative async B staging: thread -> row tid>>2 (0..63), 8-elem seg tid&3
    const int brow = tid >> 2;
    const int bseg = (tid & 3) * 8;
    const __bf16* __restrict__ bsrc = Bt + (size_t)(n0 + brow) * K + bseg;
    const uint32_t ldsOff = (uint32_t)(brow * 32 + bseg) * 2u;
    const uint32_t ldsBase0 = (uint32_t)(uintptr_t)(&ldsB[0][0]);
    const uint32_t ldsBase1 = (uint32_t)(uintptr_t)(&ldsB[1][0]);

    const __bf16* __restrict__ Arow0 = A + (size_t)(row0 + m) * lda;
    const __bf16* __restrict__ Arow1 = Arow0 + (size_t)16 * lda;

    v8f acc[2][4];
#pragma unroll
    for (int rt = 0; rt < 2; ++rt)
#pragma unroll
        for (int t = 0; t < 4; ++t)
            acc[rt][t] = (v8f){0.f,0.f,0.f,0.f,0.f,0.f,0.f,0.f};

    auto asyncB = [&](int kb, int p) {
        uint64_t g = (uint64_t)(uintptr_t)(bsrc + kb);
        uint32_t l = (p ? ldsBase1 : ldsBase0) + ldsOff;
        asm volatile("global_load_async_to_lds_b128 %0, %1, off"
                     :: "v"(l), "v"(g) : "memory");
    };
    auto loadA = [&](const __bf16* base, int kb, v16bf& a) {
        v8bf lo = *(const v8bf*)(base + kb + 8 * kh);
        v8bf hi = *(const v8bf*)(base + kb + 16 + 8 * kh);
#pragma unroll
        for (int i = 0; i < 8; ++i) { a[i] = lo[i]; a[i + 8] = hi[i]; }
    };
    auto loadB = [&](int p, int t, v16bf& b) {
        const __bf16* lp = &ldsB[p][(t * 16 + m) * 32 + 8 * kh];
        v8bf lo = *(const v8bf*)(lp);
        v8bf hi = *(const v8bf*)(lp + 16);
#pragma unroll
        for (int i = 0; i < 8; ++i) { b[i] = lo[i]; b[i + 8] = hi[i]; }
    };

    // --- prologue ---
    v16bf aCur[2], aNext[2];
    asyncB(0, 0);
    loadA(Arow0, 0, aCur[0]);
    loadA(Arow1, 0, aCur[1]);
    asm volatile("s_wait_asynccnt 0x0" ::: "memory");
    __syncthreads();

    int p = 0;
    for (int kb = 0; kb < K; kb += 32) {
        const bool last = (kb + 32 >= K);
        if (!last) {
            asyncB(kb + 32, p ^ 1);                 // next B tile -> other LDS buffer
            loadA(Arow0, kb + 32, aNext[0]);        // next A frags -> regs
            loadA(Arow1, kb + 32, aNext[1]);
        }
        // pull all 4 B fragments from LDS, then an uninterrupted 8-wmma burst
        v16bf breg[4];
#pragma unroll
        for (int t = 0; t < 4; ++t) loadB(p, t, breg[t]);
#pragma unroll
        for (int t = 0; t < 4; ++t) {
            acc[0][t] = __builtin_amdgcn_wmma_f32_16x16x32_bf16(
                            false, aCur[0], false, breg[t], (short)0, acc[0][t], false, false);
            acc[1][t] = __builtin_amdgcn_wmma_f32_16x16x32_bf16(
                            false, aCur[1], false, breg[t], (short)0, acc[1][t], false, false);
        }
        if (!last) {
            asm volatile("s_wait_asynccnt 0x0" ::: "memory");
            __syncthreads();
            aCur[0] = aNext[0];
            aCur[1] = aNext[1];
            p ^= 1;
        }
    }

    // --- epilogue ---
#pragma unroll
    for (int rt = 0; rt < 2; ++rt) {
#pragma unroll
        for (int t = 0; t < 4; ++t) {
            const int col = n0 + t * 16 + m;
#pragma unroll
            for (int r = 0; r < 8; ++r) {
                const int row = row0 + rt * 16 + r + 8 * kh;
                const size_t o = (size_t)row * ldc + col;
                float v = acc[rt][t][r];
                if (EPI == EPI_F32) {
                    ((float*)Cout)[o] = v;
                } else if (EPI == EPI_TANH_BF16) {
                    ((__bf16*)Cout)[o] = (__bf16)fast_tanh(v);
                } else if (EPI == EPI_SILU_F32) {
                    ((float*)Cout)[o] = v / (1.0f + __expf(-v));
                } else if (EPI == EPI_DECAY_F32) {
                    float w = e_vec[col] + v;
                    ((float*)Cout)[o] = __expf(-__expf(w));
                } else { // EPI_MIX_BF16
                    const size_t xo = (size_t)row * Cd + col;
                    ((__bf16*)Cout)[o] = (__bf16)(e_x[xo] + e_xx[xo] * (e_vec[col] + v));
                }
            }
        }
    }
}

// ---------------------------------------------------------------------------
// RWKV6 WKV scan: one wave32 per (b,h) head; single-wave workgroup so
// barriers are unnecessary (LDS ops are in-order per wave).  Next timestep's
// r/k/v/d loads are issued before the current step's 32-FMA body so global
// latency overlaps compute.
// ---------------------------------------------------------------------------
__global__ __launch_bounds__(32)
void wkv_scan_kernel(const float* __restrict__ r, const float* __restrict__ k,
                     const float* __restrict__ v, const float* __restrict__ d,
                     const float* __restrict__ u, float* __restrict__ y) {
    const int bh = blockIdx.x;
    const int b  = bh / NHn;
    const int h  = bh % NHn;
    const int lane = threadIdx.x;

    __shared__ float rL[32], kL[32], dL[32], uL[32];
    uL[lane] = u[h * HSn + lane];

    float s[32];
#pragma unroll
    for (int j = 0; j < 32; ++j) s[j] = 0.0f;

    const size_t base = (size_t)b * Tn * Cd + (size_t)h * HSn + lane;
    float rv = r[base], kv = k[base], vv = v[base], dv = d[base];

    for (int t = 0; t < Tn; ++t) {
        rL[lane] = rv; kL[lane] = kv; dL[lane] = dv;
        const float vcur = vv;
        if (t + 1 < Tn) {                      // software pipeline next step
            const size_t on = base + (size_t)(t + 1) * Cd;
            rv = r[on]; kv = k[on]; vv = v[on]; dv = d[on];
        }
        float acc = 0.0f;
#pragma unroll
        for (int j = 0; j < 32; ++j) {
            float kvj = kL[j] * vcur;
            acc += rL[j] * (s[j] + uL[j] * kvj);
            s[j] = dL[j] * s[j] + kvj;
        }
        y[base + (size_t)t * Cd] = acc;
    }
}

// ---------------------------------------------------------------------------
// LayerNorm over C + silu(g) gating, output bf16.
// ---------------------------------------------------------------------------
__global__ __launch_bounds__(256)
void ln_gate_kernel(const float* __restrict__ y, const float* __restrict__ g,
                    const float* __restrict__ lnw, const float* __restrict__ lnb,
                    __bf16* __restrict__ z) {
    const int t = blockIdx.x;
    const float* yr = y + (size_t)t * Cd;
    const float* gr = g + (size_t)t * Cd;

    float s = 0.0f, s2 = 0.0f;
    for (int c = threadIdx.x; c < Cd; c += 256) {
        float v = yr[c]; s += v; s2 += v * v;
    }
    __shared__ float sbuf[16];
    for (int off = 16; off; off >>= 1) {
        s  += __shfl_down(s,  off, 32);
        s2 += __shfl_down(s2, off, 32);
    }
    const int lane = threadIdx.x & 31, wv = threadIdx.x >> 5;
    if (lane == 0) { sbuf[wv] = s; sbuf[8 + wv] = s2; }
    __syncthreads();
    if (wv == 0) {
        s  = (lane < 8) ? sbuf[lane]     : 0.0f;
        s2 = (lane < 8) ? sbuf[8 + lane] : 0.0f;
        for (int off = 4; off; off >>= 1) {
            s  += __shfl_down(s,  off, 32);
            s2 += __shfl_down(s2, off, 32);
        }
        if (lane == 0) { sbuf[0] = s; sbuf[1] = s2; }
    }
    __syncthreads();
    const float mu   = sbuf[0] * (1.0f / Cd);
    const float var  = sbuf[1] * (1.0f / Cd) - mu * mu;
    const float rstd = rsqrtf(var + 1e-5f);
    for (int c = threadIdx.x; c < Cd; c += 256) {
        float yn = (yr[c] - mu) * rstd * lnw[c] + lnb[c];
        float gv = gr[c];
        float sg = gv / (1.0f + __expf(-gv));
        z[(size_t)t * Cd + c] = (__bf16)(yn * sg);
    }
}

// ---------------------------------------------------------------------------
// Host-side orchestration
// ---------------------------------------------------------------------------
extern "C" void kernel_launch(void* const* d_in, const int* in_sizes, int n_in,
                              void* d_out, int out_size, void* d_ws, size_t ws_size,
                              hipStream_t stream) {
    const float* x        = (const float*)d_in[0];
    const float* tmx      = (const float*)d_in[1];
    const float* tm_w     = (const float*)d_in[2];
    const float* tm_k     = (const float*)d_in[3];
    const float* tm_v     = (const float*)d_in[4];
    const float* tm_r     = (const float*)d_in[5];
    const float* tm_g     = (const float*)d_in[6];
    const float* w1       = (const float*)d_in[7];   // (768, 160)
    const float* w2       = (const float*)d_in[8];   // (5, 32, 768)
    const float* tdecay   = (const float*)d_in[9];   // (1,1,768)
    const float* dw1      = (const float*)d_in[10];  // (768, 64)
    const float* dw2      = (const float*)d_in[11];  // (64, 768)
    const float* u        = (const float*)d_in[12];  // (24, 32)
    const float* W_big[5] = { (const float*)d_in[13], (const float*)d_in[14],
                              (const float*)d_in[15], (const float*)d_in[16],
                              (const float*)d_in[17] };   // W_r, W_k, W_v, W_g, W_o
    const float* lnw      = (const float*)d_in[18];
    const float* lnb      = (const float*)d_in[19];
    float* out            = (float*)d_out;

    // ---- workspace layout (256B aligned slices) ----
    char* ws = (char*)d_ws;
    size_t off = 0;
    auto alloc = [&](size_t bytes) -> char* {
        char* p = ws + off;
        off += (bytes + 255) & ~(size_t)255;
        return p;
    };
    float*  xx    = (float*)alloc((size_t)Mrows * Cd * 4);
    __bf16* xmix  = (__bf16*)alloc((size_t)Mrows * Cd * 2);
    __bf16* Wt[5];
    for (int i = 0; i < 5; ++i) Wt[i] = (__bf16*)alloc((size_t)Cd * Cd * 2);
    __bf16* w1t   = (__bf16*)alloc((size_t)192 * Cd * 2);
    __bf16* w2t   = (__bf16*)alloc((size_t)5 * Cd * 32 * 2);
    __bf16* dw1t  = (__bf16*)alloc((size_t)64 * Cd * 2);
    __bf16* dw2t  = (__bf16*)alloc((size_t)Cd * 64 * 2);
    __bf16* y1    = (__bf16*)alloc((size_t)Mrows * 192 * 2);
    __bf16* xmixb[5];
    for (int i = 0; i < 5; ++i) xmixb[i] = (__bf16*)alloc((size_t)Mrows * Cd * 2);
    __bf16* t1    = (__bf16*)alloc((size_t)Mrows * 64 * 2);
    float*  rb    = (float*)alloc((size_t)Mrows * Cd * 4);
    float*  kb    = (float*)alloc((size_t)Mrows * Cd * 4);
    float*  vb    = (float*)alloc((size_t)Mrows * Cd * 4);
    float*  gb    = (float*)alloc((size_t)Mrows * Cd * 4);
    float*  db    = (float*)alloc((size_t)Mrows * Cd * 4);
    float*  yb    = (float*)alloc((size_t)Mrows * Cd * 4);
    __bf16* zb    = (__bf16*)alloc((size_t)Mrows * Cd * 2);
    (void)ws_size; (void)in_sizes; (void)n_in; (void)out_size;

    auto tgrid = [](int R, int Cpad) { return dim3((unsigned)((R * Cpad + 255) / 256)); };

    // ---- 1) weight transposes -> bf16 [N][K] ----
    for (int i = 0; i < 5; ++i)
        transpose_bf16_kernel<<<tgrid(Cd, Cd), 256, 0, stream>>>(W_big[i], Wt[i], Cd, Cd, Cd);
    transpose_bf16_kernel<<<tgrid(Cd, 192), 256, 0, stream>>>(w1, w1t, Cd, 160, 192);
    for (int kk = 0; kk < 5; ++kk)
        transpose_bf16_kernel<<<tgrid(32, Cd), 256, 0, stream>>>(
            w2 + (size_t)kk * 32 * Cd, w2t + (size_t)kk * Cd * 32, 32, Cd, Cd);
    transpose_bf16_kernel<<<tgrid(Cd, 64), 256, 0, stream>>>(dw1, dw1t, Cd, 64, 64);
    transpose_bf16_kernel<<<tgrid(64, Cd), 256, 0, stream>>>(dw2, dw2t, 64, Cd, Cd);

    // ---- 2) q_shift + maa_x mix ----
    prep_kernel<<<dim3((Mrows * Cd + 255) / 256), 256, 0, stream>>>(x, tmx, xx, xmix);

    const dim3 blk(256);
    const unsigned mg = Mrows / 256;   // 16

    // ---- 3) y1 = tanh(xmix @ w1)   (N=192, K=768) ----
    gemm_wmma_kernel<EPI_TANH_BF16><<<dim3(192 / 64, mg), blk, 0, stream>>>(
        xmix, Cd, w1t, Cd, y1, 192, nullptr, nullptr, nullptr);

    // ---- 4) mix GEMMs (K=32): x{w,k,v,r,g} = bf16(x + xx*(tm + y1[:,32k:] @ w2[k])) ----
    const float* tms[5] = { tm_w, tm_k, tm_v, tm_r, tm_g };
    for (int kk = 0; kk < 5; ++kk)
        gemm_wmma_kernel<EPI_MIX_BF16><<<dim3(Cd / 64, mg), blk, 0, stream>>>(
            y1 + 32 * kk, 192, w2t + (size_t)kk * Cd * 32, 32,
            xmixb[kk], Cd, x, xx, tms[kk]);

    // ---- 5) big projections (K=768) ----
    gemm_wmma_kernel<EPI_F32><<<dim3(Cd / 64, mg), blk, 0, stream>>>(
        xmixb[3], Cd, Wt[0], Cd, rb, Cd, nullptr, nullptr, nullptr);   // r
    gemm_wmma_kernel<EPI_F32><<<dim3(Cd / 64, mg), blk, 0, stream>>>(
        xmixb[1], Cd, Wt[1], Cd, kb, Cd, nullptr, nullptr, nullptr);   // k
    gemm_wmma_kernel<EPI_F32><<<dim3(Cd / 64, mg), blk, 0, stream>>>(
        xmixb[2], Cd, Wt[2], Cd, vb, Cd, nullptr, nullptr, nullptr);   // v
    gemm_wmma_kernel<EPI_SILU_F32><<<dim3(Cd / 64, mg), blk, 0, stream>>>(
        xmixb[4], Cd, Wt[3], Cd, gb, Cd, nullptr, nullptr, nullptr);   // g

    // decay path
    gemm_wmma_kernel<EPI_TANH_BF16><<<dim3(64 / 64, mg), blk, 0, stream>>>(
        xmixb[0], Cd, dw1t, Cd, t1, 64, nullptr, nullptr, nullptr);
    gemm_wmma_kernel<EPI_DECAY_F32><<<dim3(Cd / 64, mg), blk, 0, stream>>>(
        t1, 64, dw2t, 64, db, Cd, nullptr, nullptr, tdecay);

    // ---- 6) wkv6 scan ----
    wkv_scan_kernel<<<dim3(Bn * NHn), dim3(32), 0, stream>>>(rb, kb, vb, db, u, yb);

    // ---- 7) layernorm + silu gate -> bf16 ----
    ln_gate_kernel<<<dim3(Mrows), blk, 0, stream>>>(yb, gb, lnw, lnb, zb);

    // ---- 8) out = z @ W_o ----
    gemm_wmma_kernel<EPI_F32><<<dim3(Cd / 64, mg), blk, 0, stream>>>(
        zb, Cd, Wt[4], Cd, out, Cd, nullptr, nullptr, nullptr);
}
